// FasterRCNN_3109556322621
// MI455X (gfx1250) — compile-verified, matrix operand order checked by reading, not run on hardware
//
#include <hip/hip_runtime.h>
#include <hip/hip_bf16.h>
#include <math.h>

typedef __attribute__((ext_vector_type(16))) __bf16 v16bf;
typedef __attribute__((ext_vector_type(8)))  float  v8f;

#define NLEV 5
#define CDIM 256
#define KTOT 2304            // 9 * 256
#define NSLICE 72            // 2304 / 32
#define TOTA 242991          // total anchors per image
#define DWH_CLIPV 4.135166556742356f

#define AS1 __attribute__((address_space(1)))
#define AS3 __attribute__((address_space(3)))

#if __has_builtin(__builtin_amdgcn_global_load_async_to_lds_b128)
#define HAVE_ASYNC_LDS 1
#else
#define HAVE_ASYNC_LDS 0
#endif

#if __has_builtin(__builtin_amdgcn_tensor_load_to_lds)
#define HAVE_TDM 1
#else
#define HAVE_TDM 0
#endif

#if __has_builtin(__builtin_amdgcn_s_wait_asynccnt)
#define WAIT_ASYNC() __builtin_amdgcn_s_wait_asynccnt(0)
#elif HAVE_ASYNC_LDS
#define WAIT_ASYNC() asm volatile("s_wait_asynccnt 0x0" ::: "memory")
#else
#define WAIT_ASYNC() ((void)0)
#endif

// gcc-style int4 vector: matches the builtin's declared pointee type
typedef int i32x4_vs __attribute__((vector_size(16)));

// ---------------------------------------------------------------------------
// Pack conv weights (O=256,I=256,3,3) f32 -> bf16 in WMMA-B fragment order.
// Flat layout: (((s*16 + nt)*32 + lane)*16 + idx)
//   k = s*32 + kidx, kidx = (idx>>3)*16 + (lane>>4)*8 + (idx&7)
//   n = nt*16 + (lane&15)
//   k = tap*256 + c  (tap-major K)  => tap = s>>3, c = (s&7)*32 + kidx
// ---------------------------------------------------------------------------
__global__ __launch_bounds__(256) void pack_wp_kernel(const float* __restrict__ conv_w,
                                                      __bf16* __restrict__ Wp) {
    int i = blockIdx.x * 256 + threadIdx.x;
    if (i >= KTOT * CDIM) return;
    int idx = i & 15;
    int l   = (i >> 4) & 31;
    int nt  = (i >> 9) & 15;
    int s   = i >> 13;
    int g   = l >> 4, lc = l & 15;
    int kidx = ((idx >> 3) << 4) + (g << 3) + (idx & 7);
    int n    = nt * 16 + lc;
    int tap  = s >> 3;
    int c    = ((s & 7) << 5) + kidx;
    int dy   = tap / 3, dx = tap % 3;
    float v  = conv_w[((size_t)(n * 256 + c) * 3 + dy) * 3 + dx];
    Wp[i] = (__bf16)v;
}

// head weight table: rows 0..2 = log_w, rows 3..14 = reg_w
__global__ __launch_bounds__(256) void pack_headw_kernel(const float* __restrict__ log_w,
                                                         const float* __restrict__ reg_w,
                                                         float* __restrict__ headw) {
    int i = blockIdx.x * 256 + threadIdx.x;
    if (i >= 15 * CDIM) return;
    int o = i >> 8, c = i & 255;
    headw[i] = (o < 3) ? log_w[o * CDIM + c] : reg_w[(o - 3) * CDIM + c];
}

// ---------------------------------------------------------------------------
// Fused: conv3x3 (bf16 WMMA implicit GEMM) + bias + ReLU + 1x1 heads +
// sigmoid + box decode + clip + validity. One wave = 16 pixels x 256 ch.
// B slices stream to LDS via GLOBAL_LOAD_ASYNC_TO_LDS (ASYNCcnt);
// head weight table arrives via the Tensor Data Mover (TENSORcnt).
// ---------------------------------------------------------------------------
__global__ __launch_bounds__(256) void conv_head_kernel(
    const float* __restrict__ f, const float* __restrict__ pri,
    const __bf16* __restrict__ Wp, const float* __restrict__ headw,
    const float* __restrict__ conv_b, const float* __restrict__ log_b,
    const float* __restrict__ reg_b, const int* __restrict__ imsizes,
    float* __restrict__ scoreBuf, float* __restrict__ boxBuf,
    int H, int W, int npix, int lvOff)
{
    __shared__ __bf16 Bsh[2][8192];   // 2 x (32 K x 256 N) bf16 slices, frag order
    __shared__ float  WHsh[15 * CDIM];

    const int tid  = threadIdx.x;
    const int lane = tid & 31;
    const int wv   = tid >> 5;
    const int img  = blockIdx.y;
    const int g    = lane >> 4;
    const int lc   = lane & 15;

    // ---- stage head-weight table into LDS (TDM when available) ----
#if HAVE_TDM
    if (wv == 0) {
        typedef unsigned int v4u_t __attribute__((ext_vector_type(4)));
        typedef int          v4i_t __attribute__((ext_vector_type(4)));
        typedef int          v8i_t __attribute__((ext_vector_type(8)));
        const unsigned NB = 15 * CDIM * 4;   // 15360 bytes, one contiguous row
        unsigned lds_off = (unsigned)(unsigned long long)(AS3 char*)(void*)WHsh;
        unsigned long long ga = (unsigned long long)(uintptr_t)headw;
        v4u_t g0;
        g0[0] = 1u;                                    // count=1, user descriptor
        g0[1] = lds_off;                               // lds_addr
        g0[2] = (unsigned)(ga & 0xFFFFFFFFu);          // global_addr[31:0]
        g0[3] = (unsigned)((ga >> 32) & 0x1FFFFFFu)    // global_addr[56:32]
              | (2u << 30);                            // type=2 ("image")
        v8i_t g1;
        g1[0] = 0;                                     // wg_mask=0, data_size=1B
        g1[1] = (int)((NB & 0xFFFFu) << 16);           // tensor_dim0[15:0]
        g1[2] = (int)((NB >> 16) | (1u << 16));        // tensor_dim0[31:16] | tensor_dim1=1
        g1[3] = (int)((NB & 0xFFFFu) << 16);           // tensor_dim1 hi=0 | tile_dim0
        g1[4] = 1;                                     // tile_dim1=1, tile_dim2=0
        g1[5] = (int)NB;                               // tensor_dim0_stride[31:0]
        g1[6] = 0;                                     // stride hi, dim1_stride lo
        g1[7] = 0;
        v4i_t g2 = {0, 0, 0, 0};
        v4i_t g3 = {0, 0, 0, 0};
#if defined(__clang_major__) && (__clang_major__ >= 23)
        v8i_t g4 = {0, 0, 0, 0, 0, 0, 0, 0};
        __builtin_amdgcn_tensor_load_to_lds(g0, g1, g2, g3, g4, 0);
#else
        __builtin_amdgcn_tensor_load_to_lds(g0, g1, g2, g3, 0);
#endif
        __builtin_amdgcn_s_wait_tensorcnt(0);
    }
#else
    for (int i = tid; i < 15 * CDIM; i += 256) WHsh[i] = headw[i];
#endif

    const float* fimg = f + (size_t)img * CDIM * H * W;
    const uint4* WpV  = (const uint4*)Wp;   // 8 bf16 per uint4, 1024 uint4 per slice

    auto stage = [&](int s, int buf) {
#if HAVE_ASYNC_LDS
        AS3 char* dl = (AS3 char*)(void*)Bsh[buf];
        uintptr_t sg = (uintptr_t)WpV + ((uintptr_t)s << 14);   // 16384 B / slice
        #pragma unroll
        for (int i = 0; i < 4; ++i) {
            int off = (i * 256 + tid) * 16;
            __builtin_amdgcn_global_load_async_to_lds_b128(
                (AS1 i32x4_vs*)(sg + off), (AS3 i32x4_vs*)(dl + off), 0, 0);
        }
#else
        uint4* dst = (uint4*)Bsh[buf];
        const uint4* src = WpV + (size_t)s * 1024;
        #pragma unroll 4
        for (int i = tid; i < 1024; i += 256) dst[i] = src[i];
#endif
    };

    stage(0, 0);

    const int P = (blockIdx.x * 8 + wv) * 16;   // this wave's pixel tile base
    int p  = P + lc;
    int pc = (p < npix) ? p : (npix - 1);
    int py = pc / W, px = pc % W;

    v8f acc[16];
    #pragma unroll
    for (int nt = 0; nt < 16; ++nt)
        #pragma unroll
        for (int i = 0; i < 8; ++i) acc[nt][i] = 0.0f;

    WAIT_ASYNC();
    __syncthreads();

    const size_t cs = (size_t)H * W;
    for (int s = 0; s < NSLICE; ++s) {
        if (s + 1 < NSLICE) {
            stage(s + 1, (s + 1) & 1);
            __builtin_prefetch(WpV + (size_t)(s + 2) * 1024, 0, 1);
        }

        // ---- A fragment (im2col gather, row M = pixel, K pattern per ISA) ----
        int tap = s >> 3;
        int dy  = tap / 3 - 1, dx = tap % 3 - 1;
        int yy  = py + dy, xx = px + dx;
        bool inb = (yy >= 0) && (yy < H) && (xx >= 0) && (xx < W);
        int cb = ((s & 7) << 5) + (g << 3);
        v16bf a;
        if (inb) {
            const float* src = fimg + ((size_t)cb * H + yy) * W + xx;
            #pragma unroll
            for (int j = 0; j < 8; ++j) {
                a[j]     = (__bf16)src[(size_t)j * cs];
                a[8 + j] = (__bf16)src[(size_t)(16 + j) * cs];
            }
        } else {
            #pragma unroll
            for (int j = 0; j < 16; ++j) a[j] = (__bf16)0.0f;
        }

        // ---- 16 N-tiles of WMMA bf16 ----
        const __bf16* bp = Bsh[s & 1];
        #pragma unroll
        for (int nt = 0; nt < 16; ++nt) {
            v16bf bfrag = *(const v16bf*)(bp + ((nt * 32 + lane) << 4));
            acc[nt] = __builtin_amdgcn_wmma_f32_16x16x32_bf16(
                false, a, false, bfrag, (short)0, acc[nt], false, false);
        }
        WAIT_ASYNC();
        __syncthreads();
    }

    // ---- bias + ReLU ----
    #pragma unroll
    for (int nt = 0; nt < 16; ++nt) {
        float bs = conv_b[nt * 16 + lc];
        #pragma unroll
        for (int r = 0; r < 8; ++r)
            acc[nt][r] = fmaxf(acc[nt][r] + bs, 0.0f);
    }

    // ---- heads: 15-wide GEMV over 256 ch, cross-lane reduced, then decode ----
    const float szW = (float)imsizes[img * 2 + 1];
    const float szH = (float)imsizes[img * 2 + 0];

    #pragma unroll
    for (int r = 0; r < 8; ++r) {
        float val[15];
        for (int o = 0; o < 15; ++o) {
            float wsum = 0.0f;
            const float* wrow = WHsh + o * CDIM + lc;
            #pragma unroll
            for (int nt = 0; nt < 16; ++nt)
                wsum += acc[nt][r] * wrow[nt * 16];
            // reduce across the 16 lanes of this half (pixel = r + 8*g)
            wsum += __shfl_xor(wsum, 1, 32);
            wsum += __shfl_xor(wsum, 2, 32);
            wsum += __shfl_xor(wsum, 4, 32);
            wsum += __shfl_xor(wsum, 8, 32);
            float hb = (o < 3) ? log_b[o] : reg_b[o - 3];
            val[o] = wsum + hb;
        }
        int p2 = P + r + 8 * g;
        int a_id = lc;
        if (p2 < npix && a_id < 3) {
            const float* pr = pri + (size_t)(p2 * 3 + a_id) * 4;
            float px1 = pr[0], py1 = pr[1], px2 = pr[2], py2 = pr[3];
            float pw = px2 - px1, ph = py2 - py1;
            float pcx = px1 + 0.5f * pw, pcy = py1 + 0.5f * ph;
            float ddx = val[3 + a_id * 4 + 0];
            float ddy = val[3 + a_id * 4 + 1];
            float ddw = fminf(val[3 + a_id * 4 + 2], DWH_CLIPV);
            float ddh = fminf(val[3 + a_id * 4 + 3], DWH_CLIPV);
            float cx = pcx + ddx * pw, cy = pcy + ddy * ph;
            float bw = pw * __expf(ddw), bh = ph * __expf(ddh);
            float x1 = fminf(fmaxf(cx - 0.5f * bw, 0.0f), szW);
            float y1 = fminf(fmaxf(cy - 0.5f * bh, 0.0f), szH);
            float x2 = fminf(fmaxf(cx + 0.5f * bw, 0.0f), szW);
            float y2 = fminf(fmaxf(cy + 0.5f * bh, 0.0f), szH);
            bool valid = ((x2 - x1) > 0.001f) && ((y2 - y1) > 0.001f);
            float score = 1.0f / (1.0f + __expf(-val[a_id]));
            float rank  = valid ? score : -1.0f;
            size_t gix = (size_t)img * TOTA + lvOff + (size_t)p2 * 3 + a_id;
            scoreBuf[gix] = rank;
            float4* b4 = (float4*)boxBuf;
            b4[gix] = make_float4(x1, y1, x2, y2);
        }
    }
}

// ---------------------------------------------------------------------------
// Per-(img,level) top-k by iterative block argmax (yields sorted candidates).
// ---------------------------------------------------------------------------
__global__ __launch_bounds__(256) void topk_kernel(float* __restrict__ scoreBuf,
                                                   const float4* __restrict__ boxBuf,
                                                   float* __restrict__ candScore,
                                                   float4* __restrict__ candBox) {
    const int nTab[NLEV]   = {182400, 45600, 11400, 2850, 741};
    const int offTab[NLEV] = {0, 182400, 228000, 239400, 242250};
    const int kTab[NLEV]   = {1000, 1000, 1000, 1000, 741};
    int b = blockIdx.x;
    int img = b / NLEV, lv = b % NLEV;
    int n = nTab[lv], k = kTab[lv];
    float*        sc = scoreBuf + (size_t)img * TOTA + offTab[lv];
    const float4* bx = boxBuf + (size_t)img * TOTA + offTab[lv];
    __shared__ float sV[256];
    __shared__ int   sI[256];
    int tid = threadIdx.x;
    for (int it = 0; it < 1000; ++it) {
        if (it < k) {
            float bv = -1e30f; int bi = 0;
            for (int j = tid; j < n; j += 256) {
                float v = sc[j];
                if (v > bv) { bv = v; bi = j; }
            }
            sV[tid] = bv; sI[tid] = bi;
            __syncthreads();
            for (int o = 128; o > 0; o >>= 1) {
                if (tid < o) {
                    float v2 = sV[tid + o]; int i2 = sI[tid + o];
                    if (v2 > sV[tid] || (v2 == sV[tid] && i2 < sI[tid])) {
                        sV[tid] = v2; sI[tid] = i2;
                    }
                }
                __syncthreads();
            }
            if (tid == 0) {
                int bidx = sI[0];
                candScore[b * 1000 + it] = sV[0];
                candBox[b * 1000 + it]   = bx[bidx];
                sc[bidx] = -2.0f;   // remove from further selection
            }
            __syncthreads();
        } else if (tid == 0) {
            candScore[b * 1000 + it] = -1.0f;
            candBox[b * 1000 + it]   = make_float4(0.f, 0.f, 0.f, 0.f);
        }
    }
}

// ---------------------------------------------------------------------------
// Greedy NMS per (img,level) group (equivalent to reference's offset trick).
// ---------------------------------------------------------------------------
__global__ __launch_bounds__(256) void nms_kernel(float* __restrict__ candScore,
                                                  const float4* __restrict__ candBox) {
    const int kTab[NLEV] = {1000, 1000, 1000, 1000, 741};
    int b = blockIdx.x;
    int n = kTab[b % NLEV];
    __shared__ float4 bbx[1000];
    __shared__ float  ssc[1000];
    __shared__ int    alv[1000];
    int tid = threadIdx.x;
    for (int i = tid; i < 1000; i += 256) {
        float s = candScore[b * 1000 + i];
        ssc[i] = s;
        bbx[i] = candBox[b * 1000 + i];
        alv[i] = (s > 0.0f) ? 1 : 0;
    }
    __syncthreads();
    for (int i = 0; i < n; ++i) {
        if (alv[i]) {
            float4 bi = bbx[i];
            float ai = (bi.z - bi.x) * (bi.w - bi.y);
            for (int j = i + 1 + tid; j < n; j += 256) {
                if (alv[j]) {
                    float4 bj = bbx[j];
                    float xx1 = fmaxf(bi.x, bj.x), yy1 = fmaxf(bi.y, bj.y);
                    float xx2 = fminf(bi.z, bj.z), yy2 = fminf(bi.w, bj.w);
                    float inter = fmaxf(xx2 - xx1, 0.0f) * fmaxf(yy2 - yy1, 0.0f);
                    float aj = (bj.z - bj.x) * (bj.w - bj.y);
                    float iou = inter / (ai + aj - inter + 1e-9f);
                    if (iou > 0.7f) alv[j] = 0;
                }
            }
        }
        __syncthreads();
    }
    for (int i = tid; i < 1000; i += 256)
        candScore[b * 1000 + i] = alv[i] ? ssc[i] : -1.0f;
}

// ---------------------------------------------------------------------------
// Per-image top-1000 across the 5 surviving candidate lists -> output (2,1000,5)
// ---------------------------------------------------------------------------
__global__ __launch_bounds__(256) void final_kernel(float* __restrict__ candScore,
                                                    const float4* __restrict__ candBox,
                                                    float* __restrict__ out) {
    int img = blockIdx.x;
    int base = img * 5000;
    int tid = threadIdx.x;
    __shared__ float sV[256];
    __shared__ int   sI[256];
    for (int it = 0; it < 1000; ++it) {
        float bv = -1e30f; int bi = 0;
        for (int j = tid; j < 5000; j += 256) {
            float v = candScore[base + j];
            if (v > bv) { bv = v; bi = j; }
        }
        sV[tid] = bv; sI[tid] = bi;
        __syncthreads();
        for (int o = 128; o > 0; o >>= 1) {
            if (tid < o) {
                float v2 = sV[tid + o]; int i2 = sI[tid + o];
                if (v2 > sV[tid] || (v2 == sV[tid] && i2 < sI[tid])) {
                    sV[tid] = v2; sI[tid] = i2;
                }
            }
            __syncthreads();
        }
        if (tid == 0) {
            float v = sV[0]; int bidx = sI[0];
            float* o5 = out + ((size_t)img * 1000 + it) * 5;
            if (v > 0.0f) {
                float4 bb = candBox[base + bidx];
                o5[0] = bb.x; o5[1] = bb.y; o5[2] = bb.z; o5[3] = bb.w; o5[4] = v;
                candScore[base + bidx] = -2.0f;
            } else {
                o5[0] = 0.f; o5[1] = 0.f; o5[2] = 0.f; o5[3] = 0.f; o5[4] = 0.f;
            }
        }
        __syncthreads();
    }
}

// ---------------------------------------------------------------------------
extern "C" void kernel_launch(void* const* d_in, const int* in_sizes, int n_in,
                              void* d_out, int out_size, void* d_ws, size_t ws_size,
                              hipStream_t stream) {
    (void)in_sizes; (void)n_in; (void)out_size; (void)ws_size;
    // setup_inputs dict order: f0,p0,f1,p1,...,f4,p4, conv_w, conv_b, log_w,
    // log_b, reg_w, reg_b, imsizes
    const float* fmaps[NLEV]  = {(const float*)d_in[0], (const float*)d_in[2],
                                 (const float*)d_in[4], (const float*)d_in[6],
                                 (const float*)d_in[8]};
    const float* priors[NLEV] = {(const float*)d_in[1], (const float*)d_in[3],
                                 (const float*)d_in[5], (const float*)d_in[7],
                                 (const float*)d_in[9]};
    const float* conv_w = (const float*)d_in[10];
    const float* conv_b = (const float*)d_in[11];
    const float* log_w  = (const float*)d_in[12];
    const float* log_b  = (const float*)d_in[13];
    const float* reg_w  = (const float*)d_in[14];
    const float* reg_b  = (const float*)d_in[15];
    const int*   imsz   = (const int*)d_in[16];
    float* out = (float*)d_out;

    // workspace layout (bytes)
    char* ws = (char*)d_ws;
    __bf16* Wp        = (__bf16*)(ws + 0);            // 589824 bf16 = 1,179,648 B
    float*  headw     = (float*)(ws + 1179648);       // 3840 f32   =    15,360 B
    float*  scoreBuf  = (float*)(ws + 1195008);       // 485,982 f32 = 1,943,928 B
    float*  boxBuf    = (float*)(ws + 3138944);       // 485,982 f4 = 7,775,712 B
    float*  candScore = (float*)(ws + 10914656);      // 10,000 f32
    float4* candBox   = (float4*)(ws + 10954656);     // 10,000 f4

    // 1. pack weights
    pack_wp_kernel<<<(KTOT * CDIM + 255) / 256, 256, 0, stream>>>(conv_w, Wp);
    pack_headw_kernel<<<(15 * CDIM + 255) / 256, 256, 0, stream>>>(log_w, reg_w, headw);

    // 2. fused conv + heads + decode per level
    const int hTab[NLEV]     = {200, 100, 50, 25, 13};
    const int wTab[NLEV]     = {304, 152, 76, 38, 19};
    const int npixTab[NLEV]  = {60800, 15200, 3800, 950, 247};
    const int lvOffTab[NLEV] = {0, 182400, 228000, 239400, 242250};
    for (int lv = 0; lv < NLEV; ++lv) {
        int tiles = (npixTab[lv] + 127) / 128;
        dim3 grid(tiles, 2, 1);
        conv_head_kernel<<<grid, 256, 0, stream>>>(
            fmaps[lv], priors[lv], Wp, headw, conv_b, log_b, reg_b, imsz,
            scoreBuf, boxBuf, hTab[lv], wTab[lv], npixTab[lv], lvOffTab[lv]);
    }

    // 3. per (img,level) top-k
    topk_kernel<<<10, 256, 0, stream>>>(scoreBuf, (const float4*)boxBuf,
                                        candScore, candBox);
    // 4. grouped NMS
    nms_kernel<<<10, 256, 0, stream>>>(candScore, candBox);
    // 5. per-image top-1000 output
    final_kernel<<<2, 256, 0, stream>>>(candScore, candBox, out);
}
